// CrossModalAttentionBlock_14516989460572
// MI455X (gfx1250) — matrix operand order repaired and checked by reading, not compiled, not run on hardware
//
#include <hip/hip_runtime.h>
#include <hip/hip_bf16.h>

// ---------------------------------------------------------------------------
// CrossModalAttentionBlock, seq_len==1 ==> softmax over 1 key == 1.0.
//   e_a' = LN(e_a + e_b @ (w_out_ab @ wv_ab)^T + (w_out_ab @ bv_ab + b_out_ab))
//   e_b' = LN(e_b + e_a @ (w_out_ba @ wv_ba)^T + (w_out_ba @ bv_ba + b_out_ba))
//   attn_w_a = ones(B,1,1)
// Dominant work: two 16384x2048x2048 GEMMs -> f16 WMMA, fp32 accumulate,
// async global->LDS staging (ASYNCcnt) with double-buffered LDS.
// ---------------------------------------------------------------------------

typedef __attribute__((ext_vector_type(16))) _Float16 v16h;
typedef __attribute__((ext_vector_type(8)))  _Float16 v8h;
typedef __attribute__((ext_vector_type(4)))  _Float16 v4h;
typedef __attribute__((ext_vector_type(8)))  float    v8f;
typedef __attribute__((ext_vector_type(4)))  int      v4i;

#define BM 64
#define BN 64
#define BK 32
#define LDSTR 40   // LDS row stride in halfs (80B: 16B-aligned, bank-skewed)

// --- async global->LDS b128 copy (GLOBAL_LOAD_ASYNC_TO_LDS_B128, ASYNCcnt) --
__device__ __forceinline__ void async_copy16(const _Float16* g, _Float16* s)
{
#if __has_builtin(__builtin_amdgcn_global_load_async_to_lds_b128)
    typedef __attribute__((address_space(1))) v4i* gptr_t;
    typedef __attribute__((address_space(3))) v4i* sptr_t;
    __builtin_amdgcn_global_load_async_to_lds_b128(
        (gptr_t)(__attribute__((address_space(1))) void*)(size_t)g,
        (sptr_t)(__attribute__((address_space(3))) void*)s,
        0, 0);
#else
    unsigned laddr =
        (unsigned)(unsigned long long)(__attribute__((address_space(3))) void*)s;
    unsigned long long gaddr = (unsigned long long)(size_t)g;
    asm volatile("global_load_async_to_lds_b128 %0, %1, off"
                 :: "v"(laddr), "v"(gaddr) : "memory");
#endif
}

template <int N>
__device__ __forceinline__ void wait_async()
{
#if __has_builtin(__builtin_amdgcn_s_wait_asynccnt)
    __builtin_amdgcn_s_wait_asynccnt(N);
#else
    asm volatile("s_wait_asynccnt %0" :: "i"(N) : "memory");
#endif
}

// ---------------------------------------------------------------------------
// f16 WMMA GEMM:  out[m][n] = sum_k A[m][k] * W[n][k]   (A: MxK, W: NxK, f16)
// FUSED=true : out32[m][n] = acc + biasc[n] + resid[m][n]   (pre-LN values)
// FUSED=false: out16[m][n] = (f16)acc                        (weight fusion)
// ---------------------------------------------------------------------------
template <bool FUSED>
__global__ __launch_bounds__(128) void wmma_gemm_kernel(
    const _Float16* __restrict__ A, const _Float16* __restrict__ Wm,
    const float* __restrict__ biasc, const float* __restrict__ resid,
    float* __restrict__ out32, _Float16* __restrict__ out16,
    int M, int N, int K)
{
    __shared__ __align__(16) _Float16 As[2][BM * LDSTR];
    __shared__ __align__(16) _Float16 Bs[2][BN * LDSTR];

    const int tid   = threadIdx.x;
    const int lane  = tid & 31;
    const int wave  = tid >> 5;
    const int waveM = (wave >> 1) * 32;    // 0 / 32
    const int waveN = (wave & 1) * 32;     // 0 / 32
    const int r     = lane & 15;
    const int lo    = lane >> 4;
    const int blockM = blockIdx.x * BM;
    const int blockN = blockIdx.y * BN;

    // cooperative tile load mapping: 128 threads x 16 halfs = 64x32 tile
    const int ldrow = tid >> 1;
    const int ldseg = (tid & 1) * 16;
    const _Float16* gA = A  + (size_t)(blockM + ldrow) * K + ldseg;
    const _Float16* gB = Wm + (size_t)(blockN + ldrow) * K + ldseg;
    const int soff = ldrow * LDSTR + ldseg;

    // preload K-slab 0 into buffer 0 (4 async b128 per thread)
    async_copy16(gA,     &As[0][soff]);
    async_copy16(gA + 8, &As[0][soff + 8]);
    async_copy16(gB,     &Bs[0][soff]);
    async_copy16(gB + 8, &Bs[0][soff + 8]);

    v8f acc[2][2] = {};
    const int iters = K / BK;

    for (int it = 0; it < iters; ++it) {
        const int cur = it & 1, nxt = cur ^ 1;
        const int kn  = (it + 1) * BK;
        if (it + 1 < iters) {
            // kick off next slab into the other buffer; its previous readers
            // were closed by the end-of-iteration barrier below.
            async_copy16(gA + kn,     &As[nxt][soff]);
            async_copy16(gA + kn + 8, &As[nxt][soff + 8]);
            async_copy16(gB + kn,     &Bs[nxt][soff]);
            async_copy16(gB + kn + 8, &Bs[nxt][soff + 8]);
            wait_async<4>();   // older 4 (current buffer) complete, newest 4 fly
        } else {
            wait_async<0>();
        }
        __syncthreads();

        // A fragment (16x32 f16): lanes 0-15 row=r K{0..7,16..23}; 16-31 K{8..15,24..31}
        v16h af[2], bf[2];
#pragma unroll
        for (int mi = 0; mi < 2; ++mi) {
            const _Float16* Ar = &As[cur][(waveM + mi * 16 + r) * LDSTR];
            union { v16h v; v8h h[2]; } u;
            u.h[0] = *(const v8h*)(Ar + lo * 8);
            u.h[1] = *(const v8h*)(Ar + 16 + lo * 8);
            af[mi] = u.v;
        }
        // B fragment (32x16): column N=r per lane; lanes 0-15 K{0..7,16..23}; 16-31 K{8..15,24..31}
#pragma unroll
        for (int ni = 0; ni < 2; ++ni) {
            const _Float16* Br = &Bs[cur][(waveN + ni * 16 + r) * LDSTR];
            union { v16h v; v8h h[2]; } u;
            u.h[0] = *(const v8h*)(Br + lo * 8);
            u.h[1] = *(const v8h*)(Br + 16 + lo * 8);
            bf[ni] = u.v;
        }
#pragma unroll
        for (int mi = 0; mi < 2; ++mi)
#pragma unroll
            for (int ni = 0; ni < 2; ++ni)
                acc[mi][ni] = __builtin_amdgcn_wmma_f32_16x16x32_f16(
                    false, af[mi], false, bf[ni], (short)0, acc[mi][ni], false, false);
        __syncthreads();   // close readers of 'cur' before it is refilled
    }

    // C/D layout: VGPR v -> lanes 0-15: (M=v, N=lane); lanes 16-31: (M=8+v, N=lane-16)
#pragma unroll
    for (int mi = 0; mi < 2; ++mi) {
#pragma unroll
        for (int ni = 0; ni < 2; ++ni) {
            const int n  = blockN + waveN + ni * 16 + r;
            const int mb = blockM + waveM + mi * 16 + lo * 8;
            if constexpr (FUSED) {
                const float bc = biasc[n];
#pragma unroll
                for (int v = 0; v < 8; ++v) {
                    const size_t idx = (size_t)(mb + v) * N + n;
                    out32[idx] = acc[mi][ni][v] + bc + resid[idx];
                }
            } else {
#pragma unroll
                for (int v = 0; v < 8; ++v)
                    out16[(size_t)(mb + v) * N + n] = (_Float16)acc[mi][ni][v];
            }
        }
    }
}

// ---------------------------------------------------------------------------
// elementwise f32 -> f16 (4-wide)
// ---------------------------------------------------------------------------
__global__ void f32_to_f16_kernel(const float* __restrict__ s, _Float16* __restrict__ d, int n4)
{
    int i = blockIdx.x * blockDim.x + threadIdx.x;
    if (i < n4) {
        const float4 v = ((const float4*)s)[i];
        v4h o;
        o[0] = (_Float16)v.x; o[1] = (_Float16)v.y;
        o[2] = (_Float16)v.z; o[3] = (_Float16)v.w;
        ((v4h*)d)[i] = o;
    }
}

// ---------------------------------------------------------------------------
// transpose + convert: dst[j*n + k] = (f16) src[k*n + j]   (n x n, n % 32 == 0)
// ---------------------------------------------------------------------------
__global__ void transpose_f16_kernel(const float* __restrict__ src, _Float16* __restrict__ dst, int n)
{
    __shared__ float tile[32][33];
    const int bx = blockIdx.x * 32, by = blockIdx.y * 32;
    const int tx = threadIdx.x, ty = threadIdx.y;   // block (32, 8)
#pragma unroll
    for (int j = 0; j < 4; ++j)
        tile[ty + j * 8][tx] = src[(size_t)(by + ty + j * 8) * n + bx + tx];
    __syncthreads();
#pragma unroll
    for (int j = 0; j < 4; ++j)
        dst[(size_t)(bx + ty + j * 8) * n + by + tx] = (_Float16)tile[tx][ty + j * 8];
}

// ---------------------------------------------------------------------------
// biasc[i] = b_out[i] + sum_k w_out[i][k] * bv[k]     (tiny: 4M MAC total)
// ---------------------------------------------------------------------------
__global__ void biasc_kernel(const float* __restrict__ wo, const float* __restrict__ bv,
                             const float* __restrict__ bo, float* __restrict__ biasc, int D)
{
    int i = blockIdx.x * blockDim.x + threadIdx.x;
    if (i >= D) return;
    float s = bo[i];
    const float* row = wo + (size_t)i * D;
    for (int k = 0; k < D; ++k) s += row[k] * bv[k];
    biasc[i] = s;
}

// ---------------------------------------------------------------------------
// In-place row LayerNorm: one 256-thread block per row of D=2048
// ---------------------------------------------------------------------------
__global__ __launch_bounds__(256) void layernorm_kernel(
    float* __restrict__ y, const float* __restrict__ g, const float* __restrict__ b, int D)
{
    const int row = blockIdx.x, tid = threadIdx.x, nthr = blockDim.x;
    float* yr = y + (size_t)row * D;

    float4 v[2];
    float s = 0.f, s2 = 0.f;
#pragma unroll
    for (int i = 0; i < 2; ++i) {
        const int idx = tid + i * nthr;
        v[i] = ((const float4*)yr)[idx];
        s  += v[i].x + v[i].y + v[i].z + v[i].w;
        s2 += v[i].x * v[i].x + v[i].y * v[i].y + v[i].z * v[i].z + v[i].w * v[i].w;
    }
    __shared__ float rs[256], rq[256];
    rs[tid] = s; rq[tid] = s2;
    __syncthreads();
    for (int off = 128; off > 0; off >>= 1) {
        if (tid < off) { rs[tid] += rs[tid + off]; rq[tid] += rq[tid + off]; }
        __syncthreads();
    }
    const float mean = rs[0] / (float)D;
    const float var  = rq[0] / (float)D - mean * mean;
    const float rstd = rsqrtf(var + 1e-5f);
#pragma unroll
    for (int i = 0; i < 2; ++i) {
        const int idx = tid + i * nthr;
        const float4 gv = ((const float4*)g)[idx];
        const float4 bv = ((const float4*)b)[idx];
        float4 o;
        o.x = (v[i].x - mean) * rstd * gv.x + bv.x;
        o.y = (v[i].y - mean) * rstd * gv.y + bv.y;
        o.z = (v[i].z - mean) * rstd * gv.z + bv.z;
        o.w = (v[i].w - mean) * rstd * gv.w + bv.w;
        ((float4*)yr)[idx] = o;
    }
}

__global__ void fill_ones_kernel(float* __restrict__ p, int n)
{
    int i = blockIdx.x * blockDim.x + threadIdx.x;
    if (i < n) p[i] = 1.0f;
}

// ---------------------------------------------------------------------------
extern "C" void kernel_launch(void* const* d_in, const int* in_sizes, int n_in,
                              void* d_out, int out_size, void* d_ws, size_t ws_size,
                              hipStream_t stream)
{
    constexpr int B = 16384, D = 2048;

    const float* e_a      = (const float*)d_in[0];
    const float* e_b      = (const float*)d_in[1];
    const float* w_in_ab  = (const float*)d_in[2];
    const float* b_in_ab  = (const float*)d_in[3];
    const float* w_out_ab = (const float*)d_in[4];
    const float* b_out_ab = (const float*)d_in[5];
    const float* w_in_ba  = (const float*)d_in[6];
    const float* b_in_ba  = (const float*)d_in[7];
    const float* w_out_ba = (const float*)d_in[8];
    const float* b_out_ba = (const float*)d_in[9];
    const float* gamma_a  = (const float*)d_in[10];
    const float* beta_a   = (const float*)d_in[11];
    const float* gamma_b  = (const float*)d_in[12];
    const float* beta_b   = (const float*)d_in[13];

    // workspace layout (all 256B aligned); total ~160 MB
    char* ws = (char*)d_ws;
    const size_t ED16  = (size_t)B * D * 2;     // 67,108,864
    const size_t DD16  = (size_t)D * D * 2;     //  8,388,608
    _Float16* eA16   = (_Float16*)(ws);
    _Float16* eB16   = (_Float16*)(ws + ED16);
    _Float16* wo16   = (_Float16*)(ws + 2 * ED16);
    _Float16* wvT16  = (_Float16*)(ws + 2 * ED16 + DD16);
    _Float16* wcAB   = (_Float16*)(ws + 2 * ED16 + 2 * DD16);
    _Float16* wcBA   = (_Float16*)(ws + 2 * ED16 + 3 * DD16);
    float*    biascAB = (float*)(ws + 2 * ED16 + 4 * DD16);
    float*    biascBA = (float*)(ws + 2 * ED16 + 4 * DD16 + D * 4);

    const int nE4 = B * D / 4, nW4 = D * D / 4;

    // 1) activations -> f16
    f32_to_f16_kernel<<<(nE4 + 255) / 256, 256, 0, stream>>>(e_a, eA16, nE4);
    f32_to_f16_kernel<<<(nE4 + 255) / 256, 256, 0, stream>>>(e_b, eB16, nE4);

    const dim3 gT(D / 32, D / 32), bT(32, 8);
    const dim3 gWc(D / BM, D / BN);
    const dim3 gMain(B / BM, D / BN);

    // 2) direction a<-b: Wc_ab = w_out_ab @ wv_ab  (wv = rows [2D,3D) of w_in)
    f32_to_f16_kernel<<<(nW4 + 255) / 256, 256, 0, stream>>>(w_out_ab, wo16, nW4);
    transpose_f16_kernel<<<gT, bT, 0, stream>>>(w_in_ab + (size_t)2 * D * D, wvT16, D);
    wmma_gemm_kernel<false><<<gWc, 128, 0, stream>>>(wo16, wvT16, nullptr, nullptr,
                                                     nullptr, wcAB, D, D, D);
    biasc_kernel<<<D / 256, 256, 0, stream>>>(w_out_ab, b_in_ab + 2 * D, b_out_ab, biascAB, D);

    // 3) direction b<-a
    f32_to_f16_kernel<<<(nW4 + 255) / 256, 256, 0, stream>>>(w_out_ba, wo16, nW4);
    transpose_f16_kernel<<<gT, bT, 0, stream>>>(w_in_ba + (size_t)2 * D * D, wvT16, D);
    wmma_gemm_kernel<false><<<gWc, 128, 0, stream>>>(wo16, wvT16, nullptr, nullptr,
                                                     nullptr, wcBA, D, D, D);
    biasc_kernel<<<D / 256, 256, 0, stream>>>(w_out_ba, b_in_ba + 2 * D, b_out_ba, biascBA, D);

    // 4) main GEMMs, fused +bias +residual, pre-LN values directly into d_out
    float* yA = (float*)d_out;
    float* yB = (float*)d_out + (size_t)B * D;
    wmma_gemm_kernel<true><<<gMain, 128, 0, stream>>>(eB16, wcAB, biascAB, e_a,
                                                      yA, nullptr, B, D, D);
    wmma_gemm_kernel<true><<<gMain, 128, 0, stream>>>(eA16, wcBA, biascBA, e_b,
                                                      yB, nullptr, B, D, D);

    // 5) in-place LayerNorm
    layernorm_kernel<<<B, 256, 0, stream>>>(yA, gamma_a, beta_a, D);
    layernorm_kernel<<<B, 256, 0, stream>>>(yB, gamma_b, beta_b, D);

    // 6) attn weights: softmax over a single key == 1.0
    fill_ones_kernel<<<B / 256, 256, 0, stream>>>((float*)d_out + 2 * (size_t)B * D, B);
}
